// _GEN_NETWORK_35613868819040
// MI455X (gfx1250) — matrix operand order repaired and checked by reading, not compiled, not run on hardware
//
#include <hip/hip_runtime.h>

// ---------------------------------------------------------------- constants
static constexpr int kB  = 64;
static constexpr int kT  = 64;
static constexpr int kE  = 512;
static constexpr int kH  = 512;
static constexpr int kL  = 256;
static constexpr int kCV = 10000;
static constexpr int kFV = 500;
static constexpr int kLE = 64;
static constexpr int kLH = 128;
static constexpr int kWT = 2;

typedef __attribute__((ext_vector_type(16))) __bf16 v16bf;
typedef __attribute__((ext_vector_type(2)))  __bf16 v2bf;
typedef __attribute__((ext_vector_type(8)))  float  v8f;

// pack two fp32 -> two bf16 in one dword (lowers to v_cvt_pk_bf16_f32)
__device__ __forceinline__ unsigned pk2(float a, float b) {
    v2bf p;
    p.x = (__bf16)a;
    p.y = (__bf16)b;
    return __builtin_bit_cast(unsigned, p);
}
__device__ __forceinline__ float sigmoidf(float x) {
    return 1.0f / (1.0f + __expf(-x));
}

// ---------------------------------------------------------------- B pre-pack
// Row-major fp32 B[K,N] -> bf16 fragment-tiled layout with 128-wide n-tiles:
//   block(nt, kt) = [8 n-subtiles][32 lanes][16 bf16]  (8 KB, == LDS image)
//   n = nt*128 + sub*16 + (lane&15); k = kt*32 + (lane&16) + e
// Zero-padded to Np = ntl*128, Kp = nkt*32. Each thread emits one dword (e,e+1).
__global__ void pack_b_kernel(const float* __restrict__ src, int K, int N, int ldb,
                              __bf16* __restrict__ dst, int nkt, int ntl)
{
    long long i = (long long)blockIdx.x * blockDim.x + threadIdx.x;
    long long total = (long long)ntl * nkt * 2048;
    if (i >= total) return;
    int nt   = (int)(i / ((long long)nkt * 2048));
    int rem  = (int)(i % ((long long)nkt * 2048));
    int kt   = rem >> 11;
    int p    = (rem & 2047) << 1;
    int sub  = p >> 9;
    int lane = (p >> 4) & 31;
    int e    = p & 15;
    int n = nt * 128 + sub * 16 + (lane & 15);
    int k = kt * 32 + (lane & 16) + e;
    int nc  = n < N ? n : N - 1;
    int kc0 = k < K ? k : K - 1;
    int kc1 = (k + 1) < K ? (k + 1) : K - 1;
    float v0 = src[(long long)kc0 * ldb + nc];
    float v1 = src[(long long)kc1 * ldb + nc];
    if (n >= N || k >= K)     v0 = 0.0f;
    if (n >= N || k + 1 >= K) v1 = 0.0f;
    ((unsigned*)dst)[i] = pk2(v0, v1);
}

struct PackedB { const __bf16* p; int nkt; };

// ---------------------------------------------------------------- WMMA GEMM
// C[M,N] = A[M,K] @ B[K,N] (+bias[n]) (+row_bias[row/row_div, n])
// A fp32 row-major (converted on the fly); B pre-packed bf16 fragment tiles.
// flag_mode: 0 always write, 1 write iff flag[row]==0, 2 iff flag[row]!=0
// mask_from >= 0: columns >= mask_from forced to -inf.  Requires M % 64 == 0.
// 64x128 tile / 256 threads (8 wave32): wave (wm, wn) -> rows wm*16..+15,
// cols wn*64..+63 -> four 16x16x32 WMMAs per K-chunk. Double-buffered LDS.
template <bool KTAIL>
__global__ __launch_bounds__(256) void gemm_bf16_kernel(
    const float* __restrict__ A, int lda,
    const __bf16* __restrict__ Bp, int nkt,
    const float* __restrict__ bias,
    float* __restrict__ C, int ldc,
    int M, int N, int K,
    const float* __restrict__ row_bias, int row_div, int ld_rb,
    const int* __restrict__ flag, int flag_mode, int mask_from)
{
    __shared__ v16bf lds_a[2][4][32];   // [buf][M-subtile][lane] 16 bf16 (4 KB)
    __shared__ v16bf lds_b[2][8][32];   // [buf][N-subtile][lane] 16 bf16 (8 KB)

    const int tid  = threadIdx.x;
    const int lane = tid & 31;
    const int wave = tid >> 5;
    const int wm   = wave >> 1;          // 16-row subtile
    const int wn   = wave & 1;           // 64-col slab -> four 16x16 wmma
    const int row0 = blockIdx.y * 64;

    // A staging: thread -> row am, 8 consecutive k at akb -> 16 contiguous
    // LDS bytes of the 16-bit A fragment layout.
    const int am    = tid >> 2;
    const int akb   = (tid & 3) << 3;
    const int asub  = am >> 4;
    const int alane = (am & 15) | (akb & 16);
    const int aeb   = (akb & 8) ? 8 : 0;

    const float* Arow = A + (long long)(row0 + am) * lda + akb;
    const uint4* Bblk = (const uint4*)(Bp + (size_t)blockIdx.x * nkt * 4096);

    float ar[8];
    uint4 brq0, brq1;
    v8f c[4] = {};

    auto load_chunk = [&](int kt) {
        const int k0 = kt << 5;
        if (!KTAIL || (k0 + 32 <= K)) {
            float4 qa0 = *(const float4*)(Arow + k0);
            float4 qa1 = *(const float4*)(Arow + k0 + 4);
            ar[0] = qa0.x; ar[1] = qa0.y; ar[2] = qa0.z; ar[3] = qa0.w;
            ar[4] = qa1.x; ar[5] = qa1.y; ar[6] = qa1.z; ar[7] = qa1.w;
        } else {
#pragma unroll
            for (int j = 0; j < 8; ++j) {
                int k  = k0 + akb + j;
                int kc = k < K ? k : K - 1;                   // clamp, no branch
                float v = A[(long long)(row0 + am) * lda + kc];
                ar[j] = (k < K) ? v : 0.0f;
            }
        }
        brq0 = Bblk[kt * 512 + tid];        // packed: 8KB chunk = 512 x b128
        brq1 = Bblk[kt * 512 + 256 + tid];
    };

    auto store_lds = [&](int b) {
        uint4 qa;
        qa.x = pk2(ar[0], ar[1]);
        qa.y = pk2(ar[2], ar[3]);
        qa.z = pk2(ar[4], ar[5]);
        qa.w = pk2(ar[6], ar[7]);
        *(uint4*)((__bf16*)&lds_a[b][asub][alane] + aeb) = qa;   // ds_store_b128
        ((uint4*)&lds_b[b][0][0])[tid]       = brq0;             // ds_store_b128
        ((uint4*)&lds_b[b][0][0])[tid + 256] = brq1;             // ds_store_b128
    };

    load_chunk(0);
    store_lds(0);
    int cur = 0;
    for (int i = 0; i < nkt; ++i) {
        const bool more = (i + 1) < nkt;
        if (more) load_chunk(i + 1);          // prefetch next chunk into regs
        __syncthreads();                      // lds[cur] ready (also WAR fence)
        v16bf af = lds_a[cur][wm][lane];
#pragma unroll
        for (int s = 0; s < 4; ++s) {
            v16bf bf = lds_b[cur][wn * 4 + s][lane];
            c[s] = __builtin_amdgcn_wmma_f32_16x16x32_bf16(false, af, false, bf,
                                                           (short)0, c[s],
                                                           false, false);
        }
        if (more) store_lds(cur ^ 1);
        cur ^= 1;
    }

    // ---- epilogue: C/D layout: VGPR v -> M = wm*16 + (lane>=16 ? 8:0) + v
    const int gc0 = blockIdx.x * 128 + wn * 64 + (lane & 15);
    const int mb  = wm * 16 + ((lane >> 4) << 3);
    float bia[4] = {0.0f, 0.0f, 0.0f, 0.0f};
    if (bias) {
#pragma unroll
        for (int s = 0; s < 4; ++s)
            if (gc0 + s * 16 < N) bia[s] = bias[gc0 + s * 16];
    }
#pragma unroll
    for (int v = 0; v < 8; ++v) {
        const int row = row0 + mb + v;
        if (flag_mode == 1 && flag[row] != 0) continue;
        if (flag_mode == 2 && flag[row] == 0) continue;
        const float* rb = row_bias ? row_bias + (long long)(row / row_div) * ld_rb
                                   : (const float*)nullptr;
#pragma unroll
        for (int s = 0; s < 4; ++s) {
            const int gcol = gc0 + s * 16;
            if (gcol >= N) continue;
            float val = c[s][v] + bia[s];
            if (rb) val += rb[gcol];
            if (mask_from >= 0 && gcol >= mask_from) val = -__builtin_inff();
            C[(long long)row * ldc + gcol] = val;
        }
    }
}

// ---------------------------------------------------------------- glue kernels
__global__ void gather_rows_kernel(const float* __restrict__ table,
                                   const int* __restrict__ idx,
                                   float* __restrict__ out,
                                   int nrows, int width)
{
    long long i = (long long)blockIdx.x * blockDim.x + threadIdx.x;
    long long total = (long long)nrows * width;
    if (i >= total) return;
    int r = (int)(i / width), c = (int)(i % width);
    out[i] = table[(long long)idx[r] * width + c];
}

__global__ void add2_kernel(const float* __restrict__ a, const float* __restrict__ b,
                            float* __restrict__ o, int n)
{
    int i = blockIdx.x * blockDim.x + threadIdx.x;
    if (i < n) o[i] = a[i] + b[i];
}

__global__ void sample_z_kernel(const float* __restrict__ mu,
                                const float* __restrict__ logvar,
                                const float* __restrict__ eps,
                                float* __restrict__ z, int n)
{
    int i = blockIdx.x * blockDim.x + threadIdx.x;
    if (i < n) z[i] = __expf(0.5f * logvar[i]) * eps[i] + mu[i];
}

__global__ void step_inputs_kernel(const float* __restrict__ x_emb,
                                   const float* __restrict__ le_emb,
                                   const int* __restrict__ lseq,
                                   const float* __restrict__ uih,
                                   float* __restrict__ xcat,
                                   float* __restrict__ gru_in, int t)
{
    int i = blockIdx.x * blockDim.x + threadIdx.x;
    const int W = kLE + kE;
    if (i >= kB * W) return;
    int b = i / W, c = i % W;
    float v;
    if (c < kLE) {
        v = le_emb[((long long)b * kT + t) * kLE + c];
    } else {
        int e = c - kLE;
        float xv = x_emb[((long long)b * kT + t) * kE + e];
        v = xv;
        float lf = (float)lseq[b * kT + t];
        gru_in[b * kE + e] = xv + lf * uih[b * kE + e];
    }
    xcat[b * W + c] = v;
}

__global__ void gru_combine_kernel(const float* __restrict__ gi,
                                   const float* __restrict__ gh,
                                   float* __restrict__ h,
                                   int Bn, int Hn,
                                   float* __restrict__ hall, int t, int Tn)
{
    int i = blockIdx.x * blockDim.x + threadIdx.x;
    if (i >= Bn * Hn) return;
    int b = i / Hn, j = i % Hn;
    const float* gib = gi + (long long)b * 3 * Hn;
    const float* ghb = gh + (long long)b * 3 * Hn;
    float r  = sigmoidf(gib[j] + ghb[j]);
    float zg = sigmoidf(gib[Hn + j] + ghb[Hn + j]);
    float nn = tanhf(gib[2 * Hn + j] + r * ghb[2 * Hn + j]);
    float hv = (1.0f - zg) * nn + zg * h[i];
    h[i] = hv;
    if (hall) hall[((long long)b * Tn + t) * Hn + j] = hv;
}

__global__ void lout_kernel(const float* __restrict__ hl,
                            const float* __restrict__ lw,
                            const float* __restrict__ lb,
                            float* __restrict__ out, int t)
{
    int i = blockIdx.x * blockDim.x + threadIdx.x;
    if (i >= kB * kWT) return;
    int b = i / kWT, w = i % kWT;
    float acc = lb[w];
    for (int j = 0; j < kLH; ++j) acc += hl[b * kLH + j] * lw[j * kWT + w];
    out[((long long)b * kT + t) * kWT + w] = acc;
}

__global__ void attn_build_kernel(const float* __restrict__ h,
                                  const float* __restrict__ uh,
                                  const float* __restrict__ ih,
                                  float* __restrict__ A)
{
    int i = blockIdx.x * blockDim.x + threadIdx.x;
    const int W = kH + kL;
    if (i >= 2 * kB * W) return;
    int r = i / W, c = i % W;
    int b = r & (kB - 1);
    float v;
    if (c < kH) v = h[b * kH + c];
    else        v = (r >= kB) ? ih[b * kL + (c - kH)] : uh[b * kL + (c - kH)];
    A[i] = v;
}

__global__ void attn_score_kernel(const float* __restrict__ pre,
                                  const float* __restrict__ w2,
                                  const float* __restrict__ b2,
                                  float* __restrict__ s)
{
    int r = blockIdx.x * blockDim.x + threadIdx.x;
    if (r >= 2 * kB) return;
    float acc = b2[0];
    for (int j = 0; j < kH; ++j) acc += tanhf(pre[(long long)r * kH + j]) * w2[j];
    s[r] = acc;
}

__global__ void attn_mix_kernel(const float* __restrict__ s,
                                const float* __restrict__ uh,
                                const float* __restrict__ ih,
                                float* __restrict__ ui)
{
    int i = blockIdx.x * blockDim.x + threadIdx.x;
    if (i >= kB * kL) return;
    int b = i / kL;
    float su = s[b], si = s[kB + b];
    float m  = fmaxf(su, si);
    float eu = __expf(su - m), ei = __expf(si - m);
    float d  = eu + ei;
    ui[i] = (eu * uh[i] + ei * ih[i]) / d;
}

// ---------------------------------------------------------------- host side
static inline int blks(long long n, int t) { return (int)((n + t - 1) / t); }

static inline PackedB pack_b(hipStream_t st, __bf16*& cursor, const float* src,
                             int K, int N)
{
    const int nkt = (K + 31) / 32;
    const int ntl = (N + 127) / 128;
    __bf16* dst = cursor;
    cursor += (size_t)ntl * nkt * 4096;
    long long total = (long long)ntl * nkt * 2048;
    pack_b_kernel<<<blks(total, 256), 256, 0, st>>>(src, K, N, N, dst, nkt, ntl);
    return PackedB{dst, nkt};
}

static inline void launch_gemm(hipStream_t st,
                               const float* A, int lda, PackedB Bp,
                               const float* bias, float* C, int ldc,
                               int M, int N, int K,
                               const float* row_bias = nullptr, int row_div = 1, int ld_rb = 0,
                               const int* flag = nullptr, int flag_mode = 0, int mask_from = -1)
{
    dim3 grid((N + 127) / 128, M / 64);
    if (K & 31)
        gemm_bf16_kernel<true><<<grid, 256, 0, st>>>(
            A, lda, Bp.p, Bp.nkt, bias, C, ldc, M, N, K,
            row_bias, row_div, ld_rb, flag, flag_mode, mask_from);
    else
        gemm_bf16_kernel<false><<<grid, 256, 0, st>>>(
            A, lda, Bp.p, Bp.nkt, bias, C, ldc, M, N, K,
            row_bias, row_div, ld_rb, flag, flag_mode, mask_from);
}

extern "C" void kernel_launch(void* const* d_in, const int* in_sizes, int n_in,
                              void* d_out, int out_size, void* d_ws, size_t ws_size,
                              hipStream_t stream)
{
    (void)in_sizes; (void)n_in; (void)out_size; (void)ws_size;

    // ---- inputs (setup_inputs order)
    const int*   seq       = (const int*)  d_in[0];
    const float* input_bow = (const float*)d_in[1];
    const int*   lseq      = (const int*)  d_in[2];
    const int*   uids      = (const int*)  d_in[3];
    const int*   iids      = (const int*)  d_in[4];
    const float* eps       = (const float*)d_in[6];
    const float* emb_w     = (const float*)d_in[7];
    const float* l_emb_w   = (const float*)d_in[8];
    const float* user_emb  = (const float*)d_in[9];
    const float* item_emb  = (const float*)d_in[10];
    const float* bow_in_w  = (const float*)d_in[11];
    const float* bow_in_b  = (const float*)d_in[12];
    const float* mu_w      = (const float*)d_in[13];
    const float* mu_b      = (const float*)d_in[14];
    const float* logvar_w  = (const float*)d_in[15];
    const float* logvar_b  = (const float*)d_in[16];
    const float* mu_p_w    = (const float*)d_in[17];
    const float* mu_p_b    = (const float*)d_in[18];
    const float* logvar_p_w= (const float*)d_in[19];
    const float* logvar_p_b= (const float*)d_in[20];
    const float* lat2emb_w = (const float*)d_in[21];
    const float* lat2emb_b = (const float*)d_in[22];
    const float* gru_wi    = (const float*)d_in[23];
    const float* gru_wh    = (const float*)d_in[24];
    const float* gru_bi    = (const float*)d_in[25];
    const float* gru_bh    = (const float*)d_in[26];
    const float* grul_wi   = (const float*)d_in[27];
    const float* grul_wh   = (const float*)d_in[28];
    const float* grul_bi   = (const float*)d_in[29];
    const float* grul_bh   = (const float*)d_in[30];
    const float* func_w    = (const float*)d_in[31];
    const float* func_b    = (const float*)d_in[32];
    const float* cont_w    = (const float*)d_in[33];
    const float* cont_b    = (const float*)d_in[34];
    const float* bow2cont_w= (const float*)d_in[35];
    const float* bow2cont_b= (const float*)d_in[36];
    const float* lout_w    = (const float*)d_in[37];
    const float* lout_b    = (const float*)d_in[38];
    const float* attn_w1   = (const float*)d_in[39];
    const float* attn_b1   = (const float*)d_in[40];
    const float* attn_w2   = (const float*)d_in[41];
    const float* attn_b2   = (const float*)d_in[42];

    // ---- output offsets (floats) in return order
    float* out = (float*)d_out;
    const long long o_logits  = 0;
    const long long o_l       = (long long)kB * kT * kCV;
    const long long o_bow     = o_l + (long long)kB * kT * kWT;
    const long long o_mu      = o_bow + (long long)kB * kCV;
    const long long o_logvar  = o_mu + (long long)kB * kL;
    const long long o_mup     = o_logvar + (long long)kB * kL;
    const long long o_logvarp = o_mup + (long long)kB * kL;

    // ---- workspace carve-up: fp32 region, then bf16 packed-weight region
    float* wsf = (float*)d_ws;
    size_t off = 0;
    auto take = [&](size_t n) { float* p = wsf + off; off += n; return p; };
    float* uh      = take((size_t)kB * kL);
    float* ih      = take((size_t)kB * kL);
    float* uhih    = take((size_t)kB * kL);
    float* uih     = take((size_t)kB * kE);
    float* xe      = take((size_t)kB * kE);
    float* enc_in  = take((size_t)kB * kE);
    float* zbuf    = take((size_t)kB * kL);
    float* x_emb   = take((size_t)kB * kT * kE);
    float* le_emb  = take((size_t)kB * kT * kLE);
    float* H_all   = take((size_t)kB * kT * kH);      // row = b*T + t
    float* h       = take((size_t)kB * kH);
    float* hl      = take((size_t)kB * kLH);
    float* xcat    = take((size_t)kB * (kLE + kE));
    float* gru_in  = take((size_t)kB * kE);
    float* gi_l    = take((size_t)kB * 3 * kLH);
    float* gh_l    = take((size_t)kB * 3 * kLH);
    float* gi      = take((size_t)kB * 3 * kH);
    float* gh      = take((size_t)kB * 3 * kH);
    float* attnA   = take((size_t)2 * kB * (kH + kL));
    float* attnPre = take((size_t)2 * kB * kH);
    float* sattn   = take((size_t)2 * kB);
    float* ui      = take((size_t)kB * kL);
    __bf16* bcur = (__bf16*)(wsf + ((off + 7) & ~(size_t)7));   // 32B-align

    // ---- pre-pack every weight used as a GEMM B operand (cold, once/launch)
    PackedB P_lat2emb = pack_b(stream, bcur, lat2emb_w, kL, kE);
    PackedB P_bow_in  = pack_b(stream, bcur, bow_in_w, kCV, kE);
    PackedB P_mu      = pack_b(stream, bcur, mu_w, kE, kL);
    PackedB P_logvar  = pack_b(stream, bcur, logvar_w, kE, kL);
    PackedB P_mup     = pack_b(stream, bcur, mu_p_w, kE, kL);
    PackedB P_logvarp = pack_b(stream, bcur, logvar_p_w, kE, kL);
    PackedB P_bow2c   = pack_b(stream, bcur, bow2cont_w, kL, kCV);
    PackedB P_grulwi  = pack_b(stream, bcur, grul_wi, kLE + kE, 3 * kLH);
    PackedB P_grulwh  = pack_b(stream, bcur, grul_wh, kLH, 3 * kLH);
    PackedB P_gruwi   = pack_b(stream, bcur, gru_wi, kE, 3 * kH);
    PackedB P_gruwh   = pack_b(stream, bcur, gru_wh, kH, 3 * kH);
    PackedB P_attnw1  = pack_b(stream, bcur, attn_w1, kH + kL, kH);
    PackedB P_func    = pack_b(stream, bcur, func_w, kH, kCV);
    PackedB P_cont    = pack_b(stream, bcur, cont_w, kH, kCV);

    const int TPB = 256;

    // ---- embeddings / user-item
    gather_rows_kernel<<<blks((long long)kB * kT * kE, TPB), TPB, 0, stream>>>(
        emb_w, seq, x_emb, kB * kT, kE);
    gather_rows_kernel<<<blks((long long)kB * kT * kLE, TPB), TPB, 0, stream>>>(
        l_emb_w, lseq, le_emb, kB * kT, kLE);
    gather_rows_kernel<<<blks((long long)kB * kL, TPB), TPB, 0, stream>>>(
        user_emb, uids, uh, kB, kL);
    gather_rows_kernel<<<blks((long long)kB * kL, TPB), TPB, 0, stream>>>(
        item_emb, iids, ih, kB, kL);
    add2_kernel<<<blks(kB * kL, TPB), TPB, 0, stream>>>(uh, ih, uhih, kB * kL);

    // uih0 = (uh+ih) @ lat2emb_w + b        (64, 512, 256)
    launch_gemm(stream, uhih, kL, P_lat2emb, lat2emb_b, uih, kE, kB, kE, kL);

    // xe = input_bow @ bow_in_w + b         (64, 512, 10000)
    launch_gemm(stream, input_bow, kCV, P_bow_in, bow_in_b, xe, kE, kB, kE, kCV);

    add2_kernel<<<blks(kB * kE, TPB), TPB, 0, stream>>>(xe, uih, enc_in, kB * kE);

    launch_gemm(stream, enc_in, kE, P_mu,      mu_b,       out + o_mu,      kL, kB, kL, kE);
    launch_gemm(stream, enc_in, kE, P_logvar,  logvar_b,   out + o_logvar,  kL, kB, kL, kE);
    launch_gemm(stream, uih,    kE, P_mup,     mu_p_b,     out + o_mup,     kL, kB, kL, kE);
    launch_gemm(stream, uih,    kE, P_logvarp, logvar_p_b, out + o_logvarp, kL, kB, kL, kE);

    sample_z_kernel<<<blks(kB * kL, TPB), TPB, 0, stream>>>(
        out + o_mu, out + o_logvar, eps, zbuf, kB * kL);

    // bow_logits = z @ bow2cont + b         (64, 10000, 256)
    launch_gemm(stream, zbuf, kL, P_bow2c, bow2cont_b, out + o_bow, kCV, kB, kCV, kL);

    // ---- recurrence
    hipMemsetAsync(h,  0, (size_t)kB * kH  * sizeof(float), stream);
    hipMemsetAsync(hl, 0, (size_t)kB * kLH * sizeof(float), stream);

    for (int t = 0; t < kT; ++t) {
        step_inputs_kernel<<<blks(kB * (kLE + kE), TPB), TPB, 0, stream>>>(
            x_emb, le_emb, lseq, uih, xcat, gru_in, t);

        launch_gemm(stream, xcat, kLE + kE, P_grulwi, grul_bi, gi_l, 3 * kLH,
                    kB, 3 * kLH, kLE + kE);
        launch_gemm(stream, hl, kLH, P_grulwh, grul_bh, gh_l, 3 * kLH,
                    kB, 3 * kLH, kLH);
        gru_combine_kernel<<<blks(kB * kLH, TPB), TPB, 0, stream>>>(
            gi_l, gh_l, hl, kB, kLH, (float*)nullptr, t, kT);
        lout_kernel<<<blks(kB * kWT, 128), 128, 0, stream>>>(hl, lout_w, lout_b, out + o_l, t);

        launch_gemm(stream, gru_in, kE, P_gruwi, gru_bi, gi, 3 * kH, kB, 3 * kH, kE);
        launch_gemm(stream, h,      kH, P_gruwh, gru_bh, gh, 3 * kH, kB, 3 * kH, kH);
        gru_combine_kernel<<<blks(kB * kH, TPB), TPB, 0, stream>>>(
            gi, gh, h, kB, kH, H_all, t, kT);

        attn_build_kernel<<<blks(2 * kB * (kH + kL), TPB), TPB, 0, stream>>>(h, uh, ih, attnA);
        launch_gemm(stream, attnA, kH + kL, P_attnw1, attn_b1, attnPre, kH,
                    2 * kB, kH, kH + kL);
        attn_score_kernel<<<1, 128, 0, stream>>>(attnPre, attn_w2, attn_b2, sattn);
        attn_mix_kernel<<<blks(kB * kL, TPB), TPB, 0, stream>>>(sattn, uh, ih, ui);

        launch_gemm(stream, ui, kL, P_lat2emb, lat2emb_b, uih, kE, kB, kE, kL);
    }

    // ---- big vocab projections over all (b,t) rows: (4096, 10000, 512)
    launch_gemm(stream, H_all, kH, P_func, func_b, out + o_logits, kCV,
                kB * kT, kCV, kH,
                nullptr, 1, 0, lseq, /*flag_mode=*/1, /*mask_from=*/kFV);
    launch_gemm(stream, H_all, kH, P_cont, cont_b, out + o_logits, kCV,
                kB * kT, kCV, kH,
                out + o_bow, /*row_div=*/kT, /*ld_rb=*/kCV, lseq, /*flag_mode=*/2, -1);
}
